// MaskedCausalAttention_46205258170701
// MI455X (gfx1250) — matrix operand order repaired and checked by reading, not compiled, not run on hardware
//
#include <hip/hip_runtime.h>

#define DEVFN __device__ __forceinline__

typedef __attribute__((ext_vector_type(16))) __bf16 v16bf;
typedef __attribute__((ext_vector_type(8)))  __bf16 v8bf;
typedef __attribute__((ext_vector_type(8)))  float  v8f;
typedef __attribute__((ext_vector_type(4)))  int    v4i;

#if defined(__HIP_DEVICE_COMPILE__) && \
    __has_builtin(__builtin_amdgcn_global_load_async_to_lds_b128) && \
    __has_builtin(__builtin_amdgcn_s_wait_asynccnt)
#define HAVE_ASYNC_LDS 1
#else
#define HAVE_ASYNC_LDS 0
#endif

static constexpr int Bb = 2;
static constexpr int T  = 2048;
static constexpr int C  = 1024;
static constexpr int NH = 16;
static constexpr int Dh = 64;
static constexpr int BT = Bb * T;   // 4096

DEVFN v16bf make_v16(v8bf lo, v8bf hi) {
  return __builtin_shufflevector(lo, hi, 0,1,2,3,4,5,6,7,8,9,10,11,12,13,14,15);
}

DEVFN v8f vzero8() {
  v8f z = {0.f,0.f,0.f,0.f,0.f,0.f,0.f,0.f};
  return z;
}

// A-fragment (16 x 32 bf16, MxK): lane holds row M = lane%16,
// K chunks kb..kb+7 and kb+16..kb+23 with kb = k0 + (lane<16 ? 0 : 8)
DEVFN v16bf load_a_frag(const __bf16* __restrict__ base, int lane, int row, int ld, int k0) {
  int kb = k0 + ((lane & 16) ? 8 : 0);
  const __bf16* p = base + (size_t)row * ld + kb;
  v8bf lo = *(const v8bf*)(p);
  v8bf hi = *(const v8bf*)(p + 16);
  return make_v16(lo, hi);
}

// B-fragment (32 x 16 bf16, KxN): lane holds column N = lane%16,
// K chunks ko..ko+7 and ko+16..ko+23 with ko = k0 + (lane<16 ? 0 : 8). `colrow` is the row of the
// pre-transposed B (i.e. the output column), contiguous along K in memory.
DEVFN v16bf load_b_frag(const __bf16* __restrict__ base, int lane, int colrow, int ld, int k0) {
  int ko = k0 + ((lane & 16) ? 8 : 0);
  const __bf16* p = base + (size_t)colrow * ld + ko;
  v8bf lo = *(const v8bf*)(p);
  v8bf hi = *(const v8bf*)(p + 16);
  return make_v16(lo, hi);
}

DEVFN v8f wmma_bf16(v16bf a, v16bf b, v8f c) {
  return __builtin_amdgcn_wmma_f32_16x16x32_bf16(false, a, false, b, (short)0, c, false, false);
}

#if HAVE_ASYNC_LDS
typedef __attribute__((address_space(1))) v4i g_v4i;   // global 128-bit chunk
typedef __attribute__((address_space(3))) v4i l_v4i;   // LDS 128-bit chunk

// Stage a contiguous 4KB K block (32 keys x 64 d, bf16) into LDS:
// 8 async b128 ops, each moving 32 lanes x 16B = 512B.
DEVFN void stage_k_async(const __bf16* src, __bf16* dst, int lane) {
  #pragma unroll
  for (int i = 0; i < 8; ++i) {
    int eo = i * 256 + lane * 8;   // element offset
    __builtin_amdgcn_global_load_async_to_lds_b128(
        (g_v4i*)(src + eo), (l_v4i*)(dst + eo), 0, 0);
  }
}

// Stage a V block (64 d-rows x 32 keys, bf16) from transposed V [d, t]:
// row r is 64B at srcBase + r*T; LDS layout [64][32].
DEVFN void stage_v_async(const __bf16* srcBase, __bf16* dst, int lane) {
  #pragma unroll
  for (int i = 0; i < 8; ++i) {
    int r  = i * 8 + (lane >> 2);
    int co = (lane & 3) * 8;
    __builtin_amdgcn_global_load_async_to_lds_b128(
        (g_v4i*)(srcBase + (size_t)r * T + co),
        (l_v4i*)(dst + r * 32 + co), 0, 0);
  }
}
#endif

// ---------------- conversion kernels ----------------

__global__ void cast_f32_bf16(const float* __restrict__ in, __bf16* __restrict__ out, int n) {
  int i = blockIdx.x * blockDim.x + threadIdx.x;
  int stride = gridDim.x * blockDim.x;
  for (; i < n; i += stride) out[i] = (__bf16)in[i];
}

__global__ void transpose_to_bf16(const float* __restrict__ W, __bf16* __restrict__ WT) {
  int idx = blockIdx.x * blockDim.x + threadIdx.x;  // C*C threads
  int i = idx & (C - 1);
  int o = idx >> 10;
  if (o < C) WT[(size_t)o * C + i] = (__bf16)W[(size_t)i * C + o];
}

// ---------------- fused QKV projection ----------------
// One wave computes a 32x64 register tile of x@W (+bias): per 32-deep K step
// 2 A-fragments are reused across 4 B-fragments -> 8 WMMAs per 12 b128 loads.
// Results scatter bf16 into head-major Q/K [b,n,t,d] and transposed V [b,n,d,t].

__global__ __launch_bounds__(256) void qkv_gemm(
    const __bf16* __restrict__ xb,
    const __bf16* __restrict__ WqT, const __bf16* __restrict__ WkT, const __bf16* __restrict__ WvT,
    const float* __restrict__ bq, const float* __restrict__ bk, const float* __restrict__ bv,
    __bf16* __restrict__ Q, __bf16* __restrict__ Kc, __bf16* __restrict__ Vt) {
  int lane = threadIdx.x & 31;
  int wid  = blockIdx.x * (blockDim.x >> 5) + (threadIdx.x >> 5);
  const int MT = BT / 32;                // 128
  const int NT = C / 64;                 // 16
  const int TILES = MT * NT;             // 2048
  int mat = wid / TILES;
  int rem = wid % TILES;
  int tm = rem / NT, tn = rem % NT;

  const __bf16* Wt   = (mat == 0) ? WqT : (mat == 1) ? WkT : WvT;
  const float*  bias = (mat == 0) ? bq  : (mat == 1) ? bk  : bv;

  int m16   = lane & 15;
  int arow0 = tm * 32 + m16;
  int arow1 = arow0 + 16;

  v8f acc[2][4];
  #pragma unroll
  for (int mi = 0; mi < 2; ++mi)
    #pragma unroll
    for (int nj = 0; nj < 4; ++nj) acc[mi][nj] = vzero8();

  #pragma unroll 4
  for (int k = 0; k < C; k += 32) {
    __builtin_prefetch(xb + (size_t)arow0 * C + k + 128, 0, 1);
    __builtin_prefetch(Wt + (size_t)(tn * 64 + m16) * C + k + 128, 0, 1);
    v16bf a0 = load_a_frag(xb, lane, arow0, C, k);
    v16bf a1 = load_a_frag(xb, lane, arow1, C, k);
    #pragma unroll
    for (int nj = 0; nj < 4; ++nj) {
      v16bf b = load_b_frag(Wt, lane, tn * 64 + nj * 16 + m16, C, k);
      acc[0][nj] = wmma_bf16(a0, b, acc[0][nj]);
      acc[1][nj] = wmma_bf16(a1, b, acc[1][nj]);
    }
  }

  int half = lane >> 4;
  #pragma unroll
  for (int nj = 0; nj < 4; ++nj) {
    int ncol = tn * 64 + nj * 16 + m16;
    float bval = bias[ncol];
    int n = ncol >> 6;       // head index
    int d = ncol & 63;
    #pragma unroll
    for (int mi = 0; mi < 2; ++mi) {
      #pragma unroll
      for (int v = 0; v < 8; ++v) {
        int tg = tm * 32 + mi * 16 + v + half * 8;   // global token row
        int bz = tg >> 11;                           // / T
        int t  = tg & (T - 1);
        float val = acc[mi][nj][v] + bval;
        if (mat == 0)      Q [(((size_t)(bz * NH + n)) * T  + t) * Dh + d] = (__bf16)val;
        else if (mat == 1) Kc[(((size_t)(bz * NH + n)) * T  + t) * Dh + d] = (__bf16)val;
        else               Vt[(((size_t)(bz * NH + n)) * Dh + d) * T  + t] = (__bf16)val;
      }
    }
  }
}

// ---------------- flash attention with MGDT column mask ----------------
// One wave per (head, 16-query tile). Streams 32 keys per step:
//   K/V blocks double-buffered into LDS via GLOBAL_LOAD_ASYNC_TO_LDS_B128
//   (ASYNCcnt pipeline), 4 WMMAs for S = Q K^T, online softmax (row stats via
//   xor-shuffles inside 16-lane halves), P staged via LDS into A-fragment
//   layout, 4 WMMAs for P V.

__global__ __launch_bounds__(64) void mgdt_attention(
    const __bf16* __restrict__ Q, const __bf16* __restrict__ Kc,
    const __bf16* __restrict__ Vt, __bf16* __restrict__ O) {
  __shared__ __align__(16) __bf16 Pl[2][16 * 32];
#if HAVE_ASYNC_LDS
  __shared__ __align__(16) __bf16 Kbuf[2][2][32 * 64];   // [warp][buf][key][d]
  __shared__ __align__(16) __bf16 Vbuf[2][2][64 * 32];   // [warp][buf][d][key]
#endif
  int lane = threadIdx.x & 31;
  int warp = threadIdx.x >> 5;
  int wid  = blockIdx.x * 2 + warp;
  int head = wid >> 7;              // b*NH + n  (0..31)
  int qt   = wid & 127;
  int qbase = qt * 16;

  const __bf16* Qh = Q  + (size_t)head * T * Dh;
  const __bf16* Kh = Kc + (size_t)head * T * Dh;
  const __bf16* Vh = Vt + (size_t)head * Dh * T;
  __bf16* Pw = &Pl[warp][0];

  int m16  = lane & 15;
  int half = lane >> 4;
  int qrow = qbase + m16;

  v16bf qa0 = load_a_frag(Qh, lane, qrow, Dh, 0);
  v16bf qa1 = load_a_frag(Qh, lane, qrow, Dh, 32);

  float mrun[8], lrun[8];
  v8f oacc[4];
  #pragma unroll
  for (int v = 0; v < 8; ++v) { mrun[v] = -__builtin_inff(); lrun[v] = 0.f; }
  #pragma unroll
  for (int j = 0; j < 4; ++j) oacc[j] = vzero8();

  const float scale = 0.125f;   // 1/sqrt(64)
  const int kend = qbase + 16;

#if HAVE_ASYNC_LDS
  // prologue: stage key-block 0 into buffer 0
  stage_k_async(Kh, &Kbuf[warp][0][0], lane);
  stage_v_async(Vh, &Vbuf[warp][0][0], lane);
#endif

  for (int kb2 = 0; kb2 < kend; kb2 += 32) {
#if HAVE_ASYNC_LDS
    int buf = (kb2 >> 5) & 1;
    // issue next block's staging, then wait for current block (in-order done)
    if (kb2 + 32 < kend) {
      stage_k_async(Kh + (size_t)(kb2 + 32) * Dh, &Kbuf[warp][buf ^ 1][0], lane);
      stage_v_async(Vh + kb2 + 32, &Vbuf[warp][buf ^ 1][0], lane);
      __builtin_amdgcn_s_wait_asynccnt(16);
    } else {
      __builtin_amdgcn_s_wait_asynccnt(0);
    }
    const __bf16* Kt = &Kbuf[warp][buf][0];
    const __bf16* Vb = &Vbuf[warp][buf][0];

    v16bf kb0a = load_b_frag(Kt, lane, m16,      Dh, 0);
    v16bf kb0b = load_b_frag(Kt, lane, m16,      Dh, 32);
    v16bf kb1a = load_b_frag(Kt, lane, 16 + m16, Dh, 0);
    v16bf kb1b = load_b_frag(Kt, lane, 16 + m16, Dh, 32);
#else
    __builtin_prefetch(Kh + (size_t)(kb2 + 32 + m16) * Dh, 0, 1);
    __builtin_prefetch(Vh + (size_t)m16 * T + kb2 + 32, 0, 1);
    v16bf kb0a = load_b_frag(Kh, lane, kb2 + m16,      Dh, 0);
    v16bf kb0b = load_b_frag(Kh, lane, kb2 + m16,      Dh, 32);
    v16bf kb1a = load_b_frag(Kh, lane, kb2 + 16 + m16, Dh, 0);
    v16bf kb1b = load_b_frag(Kh, lane, kb2 + 16 + m16, Dh, 32);
#endif

    v8f Sa = vzero8(); v8f Sb = vzero8();
    Sa = wmma_bf16(qa0, kb0a, Sa);
    Sa = wmma_bf16(qa1, kb0b, Sa);
    Sb = wmma_bf16(qa0, kb1a, Sb);
    Sb = wmma_bf16(qa1, kb1b, Sb);

    int colA = kb2 + m16;
    int colB = colA + 16;
    bool mgdtA = ((colA % 3) == 1) && (colA >= 4);
    bool mgdtB = ((colB % 3) == 1) && (colB >= 4);

    // ---- online softmax per accumulator slot (row = qbase + v + 8*half)
    #pragma unroll
    for (int v = 0; v < 8; ++v) {
      int row = qbase + v + half * 8;
      float sA = (mgdtA || (colA > row)) ? -__builtin_inff() : Sa[v] * scale;
      float sB = (mgdtB || (colB > row)) ? -__builtin_inff() : Sb[v] * scale;
      float rmax = fmaxf(sA, sB);
      #pragma unroll
      for (int m = 1; m <= 8; m <<= 1) rmax = fmaxf(rmax, __shfl_xor(rmax, m, 32));
      float mnew  = fmaxf(mrun[v], rmax);
      float alpha = __expf(mrun[v] - mnew);
      float pA = __expf(sA - mnew);
      float pB = __expf(sB - mnew);
      float rsum = pA + pB;
      #pragma unroll
      for (int m = 1; m <= 8; m <<= 1) rsum += __shfl_xor(rsum, m, 32);
      lrun[v] = lrun[v] * alpha + rsum;
      mrun[v] = mnew;
      #pragma unroll
      for (int j = 0; j < 4; ++j) oacc[j][v] = oacc[j][v] * alpha;
      int prow = v + half * 8;
      Pw[prow * 32 + m16]      = (__bf16)pA;
      Pw[prow * 32 + 16 + m16] = (__bf16)pB;
    }
    __threadfence_block();   // order P stores before A-fragment reload

    // ---- reload P as A-fragment (16 queries x 32 keys)
    v16bf pa;
    {
      int kb = (lane & 16) ? 8 : 0;
      const __bf16* p = Pw + m16 * 32 + kb;
      v8bf lo = *(const v8bf*)(p);
      v8bf hi = *(const v8bf*)(p + 16);
      pa = make_v16(lo, hi);
    }

    // ---- O += P @ V  (V rows = d, contiguous along keys)
    #pragma unroll
    for (int j = 0; j < 4; ++j) {
#if HAVE_ASYNC_LDS
      v16bf vb = load_b_frag(Vb, lane, j * 16 + m16, 32, 0);
#else
      v16bf vb = load_b_frag(Vh, lane, j * 16 + m16, T, kb2);
#endif
      oacc[j] = wmma_bf16(pa, vb, oacc[j]);
    }
    __threadfence_block();   // order P reads before next iteration's stores
  }

  // ---- normalize and scatter back to [b,t,c] bf16
  int bz = head >> 4;
  int n  = head & (NH - 1);
  #pragma unroll
  for (int v = 0; v < 8; ++v) {
    float inv = 1.0f / lrun[v];
    int tg = qbase + v + half * 8;
    size_t rowoff = ((size_t)(bz * T + tg)) * C + n * Dh;
    #pragma unroll
    for (int j = 0; j < 4; ++j)
      O[rowoff + j * 16 + m16] = (__bf16)(oacc[j][v] * inv);
  }
}

// ---------------- output projection (32x64 register tile per wave) -------

__global__ __launch_bounds__(256) void proj_gemm(
    const __bf16* __restrict__ Ob, const __bf16* __restrict__ WpT,
    const float* __restrict__ bp, float* __restrict__ out) {
  int lane = threadIdx.x & 31;
  int wid  = blockIdx.x * (blockDim.x >> 5) + (threadIdx.x >> 5);
  const int NT = C / 64;                 // 16
  int tm = wid / NT, tn = wid % NT;
  int m16   = lane & 15;
  int arow0 = tm * 32 + m16;
  int arow1 = arow0 + 16;

  v8f acc[2][4];
  #pragma unroll
  for (int mi = 0; mi < 2; ++mi)
    #pragma unroll
    for (int nj = 0; nj < 4; ++nj) acc[mi][nj] = vzero8();

  #pragma unroll 4
  for (int k = 0; k < C; k += 32) {
    __builtin_prefetch(Ob + (size_t)arow0 * C + k + 128, 0, 1);
    __builtin_prefetch(WpT + (size_t)(tn * 64 + m16) * C + k + 128, 0, 1);
    v16bf a0 = load_a_frag(Ob, lane, arow0, C, k);
    v16bf a1 = load_a_frag(Ob, lane, arow1, C, k);
    #pragma unroll
    for (int nj = 0; nj < 4; ++nj) {
      v16bf b = load_b_frag(WpT, lane, tn * 64 + nj * 16 + m16, C, k);
      acc[0][nj] = wmma_bf16(a0, b, acc[0][nj]);
      acc[1][nj] = wmma_bf16(a1, b, acc[1][nj]);
    }
  }

  int half = lane >> 4;
  #pragma unroll
  for (int nj = 0; nj < 4; ++nj) {
    int ncol = tn * 64 + nj * 16 + m16;
    float bval = bp[ncol];
    #pragma unroll
    for (int mi = 0; mi < 2; ++mi) {
      #pragma unroll
      for (int v = 0; v < 8; ++v) {
        int tg = tm * 32 + mi * 16 + v + half * 8;
        out[(size_t)tg * C + ncol] = acc[mi][nj][v] + bval;
      }
    }
  }
}

// ---------------- launcher ----------------

extern "C" void kernel_launch(void* const* d_in, const int* in_sizes, int n_in,
                              void* d_out, int out_size, void* d_ws, size_t ws_size,
                              hipStream_t stream) {
  (void)in_sizes; (void)n_in; (void)out_size; (void)ws_size;
  const float* x  = (const float*)d_in[0];
  const float* Wq = (const float*)d_in[1];
  const float* bq = (const float*)d_in[2];
  const float* Wk = (const float*)d_in[3];
  const float* bk = (const float*)d_in[4];
  const float* Wv = (const float*)d_in[5];
  const float* bv = (const float*)d_in[6];
  const float* Wp = (const float*)d_in[7];
  const float* bp = (const float*)d_in[8];
  float* out = (float*)d_out;

  char* ws = (char*)d_ws;
  const size_t MB = 1024 * 1024;
  __bf16* xb  = (__bf16*)(ws + 0);        // 8 MB  : x bf16 [BT, C]
  __bf16* WqT = (__bf16*)(ws + 8 * MB);   // 2 MB  : Wq^T bf16 [C, C]
  __bf16* WkT = (__bf16*)(ws + 10 * MB);
  __bf16* WvT = (__bf16*)(ws + 12 * MB);
  __bf16* WpT = (__bf16*)(ws + 14 * MB);
  __bf16* Q   = (__bf16*)(ws + 16 * MB);  // 8 MB  : [b,n,t,d]
  __bf16* Kc  = (__bf16*)(ws + 24 * MB);  // 8 MB  : [b,n,t,d]
  __bf16* Vt  = (__bf16*)(ws + 32 * MB);  // 8 MB  : [b,n,d,t]
  __bf16* Ob  = (__bf16*)(ws + 40 * MB);  // 8 MB  : attn out bf16 [BT, C]

  cast_f32_bf16<<<4096, 256, 0, stream>>>(x, xb, BT * C);
  transpose_to_bf16<<<4096, 256, 0, stream>>>(Wq, WqT);
  transpose_to_bf16<<<4096, 256, 0, stream>>>(Wk, WkT);
  transpose_to_bf16<<<4096, 256, 0, stream>>>(Wv, WvT);
  transpose_to_bf16<<<4096, 256, 0, stream>>>(Wp, WpT);

  // 3 * (4096/32) * (1024/64) = 6144 wave-tiles, 8 waves / block
  qkv_gemm<<<768, 256, 0, stream>>>(xb, WqT, WkT, WvT, bq, bk, bv, Q, Kc, Vt);

  // 32 heads * 128 query tiles = 4096 wave tasks, 2 waves / block
  mgdt_attention<<<2048, 64, 0, stream>>>(Q, Kc, Vt, Ob);

  // (4096/32) * (1024/64) = 2048 wave-tiles, 8 waves / block
  proj_gemm<<<256, 256, 0, stream>>>(Ob, WpT, bp, out);
}
